// VarianceAdaptor_40664750359347
// MI455X (gfx1250) — compile-verified, matrix operand order checked by reading, not run on hardware
//
#include <hip/hip_runtime.h>
#include <hip/hip_bf16.h>

// ---------------------------------------------------------------------------
// CDNA5 (gfx1250) VarianceAdaptor.
// All dense matrix math goes through v_wmma_f32_16x16x32_f16 (f16 operands,
// f32 accumulate).  wave32 only; every GEMM here has M % 128 == 0 (4 waves of
// 32 rows per block), N % 32 == 0, K % 32 == 0, so the WMMA kernel has no
// edge handling and EXEC is all-ones at every WMMA (ISA requirement).
// Scratch: bump-allocated from d_ws (~0.5 GB peak; duration-branch scratch is
// reused by the pitch branch).
// ---------------------------------------------------------------------------

typedef __attribute__((ext_vector_type(16))) _Float16 v16h;
typedef __attribute__((ext_vector_type(8)))  float    v8f;

// -------------------------- WMMA GEMM kernel -------------------------------
// C[M,N] (f32) = alpha * A[M,K] (f16, row-major, lda) * Bt[N,K]^T (f16, row-
// major rows are columns of B, ldb)  (+ bias[N] | += existing C)
// Fragment layout for 16-bit A 16x32 (ISA 7.12.2): lane L holds row (L&15);
// lanes 0-15 carry K {0..7, 16..23}, lanes 16-31 carry K {8..15, 24..31}.
// B mirrors A with N taking the role of M, so a Bt row load is identical.
__device__ __forceinline__ v16h load_frag16(const _Float16* p) {
  union { v16h h; uint4 u[2]; } f;
  const uint4* q = reinterpret_cast<const uint4*>(p);
  f.u[0] = q[0];   // halves k .. k+7
  f.u[1] = q[2];   // halves k+16 .. k+23
  return f.h;
}

__global__ __launch_bounds__(128) void wmma_gemm_f16(
    const _Float16* __restrict__ A, const _Float16* __restrict__ Bt,
    float* __restrict__ C, const float* __restrict__ bias,
    int K, int lda, int ldb, int ldc,
    long sA1, long sA2, long sB1, long sB2, long sC1, long sC2,
    int zdiv, float alpha, int accum)
{
  const int z  = blockIdx.z;
  const int z1 = z / zdiv, z2 = z % zdiv;
  A  += z1 * sA1 + z2 * sA2;
  Bt += z1 * sB1 + z2 * sB2;
  C  += z1 * sC1 + z2 * sC2;

  const int lane  = threadIdx.x;
  const int tileM = (blockIdx.x * 4 + threadIdx.y) * 32;
  const int tileN = blockIdx.y * 32;
  const int rsel  = (lane & 16) ? 8 : 0;

  const _Float16* a0p = A  + (long)(tileM + (lane & 15)) * lda + rsel;
  const _Float16* a1p = a0p + (long)16 * lda;
  const _Float16* b0p = Bt + (long)(tileN + (lane & 15)) * ldb + rsel;
  const _Float16* b1p = b0p + (long)16 * ldb;

  v8f acc00{}, acc01{}, acc10{}, acc11{};
  for (int k0 = 0; k0 < K; k0 += 32) {
    v16h a0 = load_frag16(a0p + k0);
    v16h a1 = load_frag16(a1p + k0);
    v16h b0 = load_frag16(b0p + k0);
    v16h b1 = load_frag16(b1p + k0);
    if (k0 + 32 < K) {            // global_prefetch_b8 of next K block
      __builtin_prefetch(a0p + k0 + 32, 0, 1);
      __builtin_prefetch(b0p + k0 + 32, 0, 1);
    }
    acc00 = __builtin_amdgcn_wmma_f32_16x16x32_f16(false, a0, false, b0, (short)0, acc00, false, false);
    acc01 = __builtin_amdgcn_wmma_f32_16x16x32_f16(false, a0, false, b1, (short)0, acc01, false, false);
    acc10 = __builtin_amdgcn_wmma_f32_16x16x32_f16(false, a1, false, b0, (short)0, acc10, false, false);
    acc11 = __builtin_amdgcn_wmma_f32_16x16x32_f16(false, a1, false, b1, (short)0, acc11, false, false);
  }

  // C/D layout: lanes 0-15 -> col=lane, rows r (VGPR r); lanes 16-31 -> col=lane-16, rows r+8
  const int r0  = (lane >> 4) << 3;
  const int col = lane & 15;
  for (int r = 0; r < 8; ++r) {
    const int m0 = tileM + r0 + r, m1 = m0 + 16;
    const int n0 = tileN + col,    n1 = n0 + 16;
    float v00 = alpha * acc00[r], v01 = alpha * acc01[r];
    float v10 = alpha * acc10[r], v11 = alpha * acc11[r];
    if (accum) {
      C[(long)m0 * ldc + n0] += v00;  C[(long)m0 * ldc + n1] += v01;
      C[(long)m1 * ldc + n0] += v10;  C[(long)m1 * ldc + n1] += v11;
    } else {
      float b0v = bias ? bias[n0] : 0.f;
      float b1v = bias ? bias[n1] : 0.f;
      C[(long)m0 * ldc + n0] = v00 + b0v;  C[(long)m0 * ldc + n1] = v01 + b1v;
      C[(long)m1 * ldc + n0] = v10 + b0v;  C[(long)m1 * ldc + n1] = v11 + b1v;
    }
  }
}

// -------------------------- helper kernels ---------------------------------
__global__ void cvt16_k(const float* __restrict__ s, _Float16* __restrict__ d, long n) {
  long i = (long)blockIdx.x * 256 + threadIdx.x;
  if (i < n) d[i] = (_Float16)s[i];
}

// dst[c*Cin+d] = w[(c*Cin+d)*Kt + t]   (extract conv tap t from [C,Cin,K])
__global__ void tap16_k(const float* __restrict__ w, _Float16* __restrict__ d, int n, int Kt, int t) {
  int i = blockIdx.x * 256 + threadIdx.x;
  if (i < n) d[i] = (_Float16)w[(long)i * Kt + t];
}

__global__ void slice16_k(const float* __restrict__ s, _Float16* __restrict__ d,
                          long rows, int cols, int srcLd, int srcOff) {
  long i = (long)blockIdx.x * 256 + threadIdx.x;
  if (i >= rows * (long)cols) return;
  long r = i / cols; int c = (int)(i % cols);
  d[i] = (_Float16)s[r * srcLd + srcOff + c];
}

// Vt[b][h][dh][n] = KV[b*Nk+n][srcOff + h*64 + dh]  (pre-transpose V for PV GEMM)
__global__ void vtrans16_k(const float* __restrict__ kv, _Float16* __restrict__ d,
                           int Bn, int Nk, int srcLd, int srcOff) {
  long i = (long)blockIdx.x * 256 + threadIdx.x;
  long total = (long)Bn * 8 * 64 * Nk;
  if (i >= total) return;
  int n = (int)(i % Nk); long r = i / Nk;
  int dd = (int)(r % 64); long bh = r / 64;
  int h = (int)(bh % 8), b = (int)(bh / 8);
  d[i] = (_Float16)kv[((long)b * Nk + n) * srcLd + srcOff + h * 64 + dd];
}

// f32 [B,L,D] -> f16 [B,L+2,D] with zero rows 0 and L+1 (conv SAME padding)
__global__ void pad16_k(const float* __restrict__ s, _Float16* __restrict__ d,
                        int Bn, int L, int Dw) {
  long i = (long)blockIdx.x * 256 + threadIdx.x;
  long total = (long)Bn * (L + 2) * Dw;
  if (i >= total) return;
  int c = (int)(i % Dw);
  long rr = i / Dw;
  int row = (int)(rr % (L + 2));
  int b = (int)(rr / (L + 2));
  d[i] = (row == 0 || row == L + 1) ? (_Float16)0.f
                                    : (_Float16)s[((long)b * L + row - 1) * Dw + c];
}

__global__ void zero_rows16_k(_Float16* __restrict__ d, int Bn, int L, int Dw) {
  int i = blockIdx.x * 256 + threadIdx.x;
  int total = Bn * 2 * Dw;
  if (i >= total) return;
  int c = i % Dw; int r = (i / Dw) % 2; int b = i / (2 * Dw);
  d[((long)b * (L + 2) + (r ? (L + 1) : 0)) * Dw + c] = (_Float16)0.f;
}

// relu -> layernorm over 256 channels; optional f16 padded out + f32 out
__global__ __launch_bounds__(256) void relu_ln_k(
    const float* __restrict__ src, const float* __restrict__ g, const float* __restrict__ bb,
    _Float16* __restrict__ out16pad, float* __restrict__ out32, int L) {
  const int Cw = 256;
  int c = threadIdx.x, l = blockIdx.x, b = blockIdx.y;
  long row = (long)b * L + l;
  float x = fmaxf(src[row * Cw + c], 0.f);
  __shared__ float s1[256], s2[256];
  s1[c] = x; s2[c] = x * x; __syncthreads();
  for (int o = 128; o > 0; o >>= 1) {
    if (c < o) { s1[c] += s1[c + o]; s2[c] += s2[c + o]; }
    __syncthreads();
  }
  float mean = s1[0] * (1.f / Cw);
  float var  = s2[0] * (1.f / Cw) - mean * mean;
  float y = (x - mean) * rsqrtf(var + 1e-5f) * g[c] + bb[c];
  if (out16pad) out16pad[((long)b * (L + 2) + l + 1) * Cw + c] = (_Float16)y;
  if (out32)    out32[row * Cw + c] = y;
}

__global__ __launch_bounds__(256) void softmax16_k(const float* __restrict__ src,
                                                   _Float16* __restrict__ dst) {
  const int W = 256;
  long row = blockIdx.x;
  int c = threadIdx.x;
  float v = src[row * W + c];
  __shared__ float s[256];
  s[c] = v; __syncthreads();
  for (int o = 128; o > 0; o >>= 1) { if (c < o) s[c] = fmaxf(s[c], s[c + o]); __syncthreads(); }
  float mx = s[0]; __syncthreads();
  float e = __expf(v - mx);
  s[c] = e; __syncthreads();
  for (int o = 128; o > 0; o >>= 1) { if (c < o) s[c] += s[c + o]; __syncthreads(); }
  dst[row * W + c] = (_Float16)(e / s[0]);
}

// out[row] = dot(src[row,:256], w) + bias[0]  (one wave32 per row)
__global__ void rowdot_k(const float* __restrict__ src, const float* __restrict__ w,
                         const float* __restrict__ bias, float* __restrict__ out, long R) {
  const int Cw = 256;
  int lane = threadIdx.x;
  long row = (long)blockIdx.x * 4 + threadIdx.y;
  if (row >= R) return;
  float s = 0.f;
  for (int k = lane; k < Cw; k += 32) s += src[row * Cw + k] * w[k];
  for (int o = 16; o > 0; o >>= 1) s += __shfl_xor(s, o, 32);
  if (lane == 0) out[row] = s + bias[0];
}

__global__ void init_t_k(int* pT) { *pT = 0; }

// per-batch inclusive scan of clamped durations; T = max over b of totals
__global__ __launch_bounds__(256) void scan_k(const int* __restrict__ td, int* __restrict__ cs,
                                              int* __restrict__ pT) {
  const int Nn = 256;
  int b = blockIdx.x, i = threadIdx.x;
  __shared__ int s[256];
  int d = td[b * Nn + i]; if (d < 1) d = 1;
  s[i] = d; __syncthreads();
  for (int o = 1; o < Nn; o <<= 1) {
    int v = (i >= o) ? s[i - o] : 0;
    __syncthreads();
    s[i] += v;
    __syncthreads();
  }
  cs[b * Nn + i] = s[i];
  if (i == Nn - 1) atomicMax(pT, s[i]);
}

// gather/expand phonemes into combined[:, 0:512] (f16), zero beyond total
__global__ __launch_bounds__(256) void expand_k(const float* __restrict__ phon,
                                                const int* __restrict__ cs,
                                                _Float16* __restrict__ comb,
                                                int Tmax) {
  const int Nn = 256, Dw = 512, ldc = 768;
  int t = blockIdx.x, b = blockIdx.y, c = threadIdx.x;
  const int* csb = cs + b * Nn;
  int total = csb[Nn - 1];
  long orow = ((long)b * Tmax + t) * ldc;
  if (t >= total) {
    comb[orow + c] = (_Float16)0.f;
    comb[orow + c + 256] = (_Float16)0.f;
    return;
  }
  int lo = 0, hi = Nn;                       // searchsorted(cs, t, 'right')
  while (lo < hi) { int mid = (lo + hi) >> 1; if (csb[mid] <= t) lo = mid + 1; else hi = mid; }
  int idx = lo < Nn - 1 ? lo : Nn - 1;
  long srow = ((long)b * Nn + idx) * Dw;
  comb[orow + c]       = (_Float16)phon[srow + c];
  comb[orow + c + 256] = (_Float16)phon[srow + c + 256];
}

// predicted_duration -> d_out[B*T*512 + i]
__global__ void dur_out_k(const float* __restrict__ raw, const int* __restrict__ pT,
                          float* __restrict__ outp) {
  const int total = 8 * 256;
  int i = blockIdx.x * 256 + threadIdx.x;
  if (i >= total) return;
  float x = raw[i];
  float sp = (x > 20.f) ? x : log1pf(__expf(x));
  float v = fminf(fmaxf(sp + 0.5f, 1.f), 100.f);
  int T = *pT;
  outp[(long)8 * T * 512 + i] = v;
}

// pitch_emb into combined[:,512:768] (f16) + predicted_pitch output
__global__ __launch_bounds__(256) void pitch_fill_k(
    const float* __restrict__ praw, const float* __restrict__ pe_w,
    const float* __restrict__ pe_b, _Float16* __restrict__ comb,
    float* __restrict__ outp, const int* __restrict__ pT, int Tmax) {
  int t = blockIdx.x, b = blockIdx.y, j = threadIdx.x;
  float p = praw[(long)b * Tmax + t];
  long row = ((long)b * Tmax + t) * 768;
  comb[row + 512 + j] = (_Float16)(p * pe_w[j] + pe_b[j]);
  int T = *pT;
  if (j == 0 && t < T) outp[(long)8 * T * 512 + 8 * 256 + (long)b * T + t] = p;
}

// p_c rows t < T -> d_out
__global__ __launch_bounds__(256) void pc_out_k(const float* __restrict__ pc,
                                                const int* __restrict__ pT,
                                                float* __restrict__ outp, int Tmax) {
  int t = blockIdx.x, b = blockIdx.y, c = threadIdx.x;
  int T = *pT;
  if (t >= T) return;
  long s = ((long)b * Tmax + t) * 512;
  long d = ((long)b * T + t) * 512;
  outp[d + c]       = pc[s + c];
  outp[d + c + 256] = pc[s + c + 256];
}

// -------------------------- host orchestration -----------------------------
static inline void launch_gemm(hipStream_t st, const _Float16* A, const _Float16* Bt,
                               float* C, const float* bias, int M, int N, int K,
                               int lda, int ldb, int ldc, int Z = 1, int zdiv = 1,
                               long sA1 = 0, long sA2 = 0, long sB1 = 0, long sB2 = 0,
                               long sC1 = 0, long sC2 = 0, float alpha = 1.f, int accum = 0) {
  dim3 g(M / 128, N / 32, Z), b(32, 4, 1);
  wmma_gemm_f16<<<g, b, 0, st>>>(A, Bt, C, bias, K, lda, ldb, ldc,
                                 sA1, sA2, sB1, sB2, sC1, sC2, zdiv, alpha, accum);
}

extern "C" void kernel_launch(void* const* d_in, const int* in_sizes, int n_in,
                              void* d_out, int out_size, void* d_ws, size_t ws_size,
                              hipStream_t stream) {
  (void)in_sizes; (void)n_in; (void)out_size; (void)ws_size;
  const float* phon     = (const float*)d_in[0];
  const float* spk      = (const float*)d_in[1];
  const int*   tdur     = (const int*)  d_in[2];
  const float* da_in_w  = (const float*)d_in[3];
  const float* da_in_b  = (const float*)d_in[4];
  const float* da_out_w = (const float*)d_in[5];
  const float* da_out_b = (const float*)d_in[6];
  const float* pa_in_w  = (const float*)d_in[7];
  const float* pa_in_b  = (const float*)d_in[8];
  const float* pa_out_w = (const float*)d_in[9];
  const float* pa_out_b = (const float*)d_in[10];
  const float* dp_c1_w  = (const float*)d_in[11];
  const float* dp_c1_b  = (const float*)d_in[12];
  const float* dp_ln1_g = (const float*)d_in[13];
  const float* dp_ln1_b = (const float*)d_in[14];
  const float* dp_c2_w  = (const float*)d_in[15];
  const float* dp_c2_b  = (const float*)d_in[16];
  const float* dp_ln2_g = (const float*)d_in[17];
  const float* dp_ln2_b = (const float*)d_in[18];
  const float* dp_o_w   = (const float*)d_in[19];
  const float* dp_o_b   = (const float*)d_in[20];
  const float* pp_c1_w  = (const float*)d_in[21];
  const float* pp_c1_b  = (const float*)d_in[22];
  const float* pp_ln1_g = (const float*)d_in[23];
  const float* pp_ln1_b = (const float*)d_in[24];
  const float* pp_c2_w  = (const float*)d_in[25];
  const float* pp_c2_b  = (const float*)d_in[26];
  const float* pp_ln2_g = (const float*)d_in[27];
  const float* pp_ln2_b = (const float*)d_in[28];
  const float* pp_o_w   = (const float*)d_in[29];
  const float* pp_o_b   = (const float*)d_in[30];
  const float* pe_w     = (const float*)d_in[31];
  const float* pe_b     = (const float*)d_in[32];
  const float* proj_w   = (const float*)d_in[33];
  const float* proj_b   = (const float*)d_in[34];
  float* out = (float*)d_out;

  const int B = 8, Nn = 256, Dd = 512, Cc = 256;
  const int Tmax = 2048;                 // durations clamp to [1,8] -> T <= 256*8
  const long BN = (long)B * Nn;          // 2048 rows
  const long BT = (long)B * Tmax;        // 16384 rows

  char* base = (char*)d_ws;
  size_t cur = 0;
  auto alloc = [&](size_t bytes) -> void* {
    void* p = base + cur;
    cur = (cur + bytes + 255) & ~(size_t)255;
    return p;
  };
  auto A16 = [&](size_t n) { return (_Float16*)alloc(n * sizeof(_Float16)); };
  auto A32 = [&](size_t n) { return (float*)alloc(n * sizeof(float)); };
  auto cvt = [&](const float* s, _Float16* d, long n) {
    cvt16_k<<<dim3((unsigned)((n + 255) / 256)), 256, 0, stream>>>(s, d, n);
  };

  // ---- persistent region ----
  int* Tg = (int*)alloc(16);
  int* cs = (int*)alloc(BN * sizeof(int));
  _Float16 *Wdaq = A16(512 * 512), *Wdakv = A16(1024 * 512), *Wdaout = A16(512 * 512);
  _Float16 *Wpaq = A16(512 * 512), *Wpakv = A16(1024 * 512), *Wpaout = A16(512 * 512);
  _Float16 *Wdpc1[3], *Wdpc2[3], *Wppc1[3], *Wppc2[3];
  for (int t = 0; t < 3; ++t) { Wdpc1[t] = A16(256 * 512); Wdpc2[t] = A16(256 * 256); }
  for (int t = 0; t < 3; ++t) { Wppc1[t] = A16(256 * 512); Wppc2[t] = A16(256 * 256); }
  _Float16* Wproj  = A16(512 * 768);
  _Float16* phon16 = A16(BN * Dd);
  _Float16* spk16  = A16(BN * Dd);
  _Float16* comb16 = A16(BT * 768);      // [expanded(512) | pitch_emb(256)] f16
  float* praw = A32(BT);
  float* draw = A32(BN);
  size_t chk = cur;                      // scratch below is reused per branch

  // ---- weight / input conversions, duration scan ----
  cvt(da_in_w, Wdaq, 512 * 512);
  cvt(da_in_w + 512 * 512, Wdakv, 1024 * 512);
  cvt(da_out_w, Wdaout, 512 * 512);
  cvt(pa_in_w, Wpaq, 512 * 512);
  cvt(pa_in_w + 512 * 512, Wpakv, 1024 * 512);
  cvt(pa_out_w, Wpaout, 512 * 512);
  cvt(proj_w, Wproj, 512 * 768);
  for (int t = 0; t < 3; ++t) {
    tap16_k<<<(256 * 512 + 255) / 256, 256, 0, stream>>>(dp_c1_w, Wdpc1[t], 256 * 512, 3, t);
    tap16_k<<<(256 * 256 + 255) / 256, 256, 0, stream>>>(dp_c2_w, Wdpc2[t], 256 * 256, 3, t);
    tap16_k<<<(256 * 512 + 255) / 256, 256, 0, stream>>>(pp_c1_w, Wppc1[t], 256 * 512, 3, t);
    tap16_k<<<(256 * 256 + 255) / 256, 256, 0, stream>>>(pp_c2_w, Wppc2[t], 256 * 256, 3, t);
  }
  cvt(phon, phon16, BN * Dd);
  cvt(spk, spk16, BN * Dd);
  init_t_k<<<1, 1, 0, stream>>>(Tg);
  scan_k<<<B, 256, 0, stream>>>(tdur, cs, Tg);

  // =================== duration branch (Nq = 256) =========================
  {
    cur = chk;
    float*     Q32  = A32(BN * 512);
    _Float16*  Q16  = A16(BN * 512);
    float*     KV32 = A32(BN * 1024);
    _Float16*  K16  = A16(BN * 512);
    _Float16*  Vt16 = A16(BN * 512);
    float*     S32  = A32((long)64 * 256 * 256);
    _Float16*  P16  = A16((long)64 * 256 * 256);
    float*     O32  = A32(BN * 512);
    _Float16*  O16  = A16(BN * 512);
    float*     dq32 = A32(BN * 512);
    _Float16*  dqp  = A16((long)B * 258 * 512);
    float*     h1   = A32(BN * 256);
    _Float16*  h1p  = A16((long)B * 258 * 256);
    float*     h2   = A32(BN * 256);
    float*     h2n  = A32(BN * 256);

    launch_gemm(stream, phon16, Wdaq, Q32, da_in_b, 2048, 512, 512, 512, 512, 512);
    cvt(Q32, Q16, BN * 512);
    launch_gemm(stream, spk16, Wdakv, KV32, da_in_b + 512, 2048, 1024, 512, 512, 512, 1024);
    slice16_k<<<(unsigned)((BN * 512 + 255) / 256), 256, 0, stream>>>(KV32, K16, BN, 512, 1024, 0);
    vtrans16_k<<<(unsigned)((BN * 512 + 255) / 256), 256, 0, stream>>>(KV32, Vt16, B, 256, 1024, 512);
    // scores: per (b,h): Q[256x64] x K^T  -> S[b,h,256,256], *1/sqrt(64)
    launch_gemm(stream, Q16, K16, S32, nullptr, 256, 256, 64, 512, 512, 256,
                64, 8, 131072, 64, 131072, 64, 524288, 65536, 0.125f);
    softmax16_k<<<64 * 256, 256, 0, stream>>>(S32, P16);
    // PV: P[256x256] x Vt^T[256x64] -> O written as [b,n,h*64+d]
    launch_gemm(stream, P16, Vt16, O32, nullptr, 256, 64, 256, 256, 256, 512,
                64, 8, 524288, 65536, 131072, 16384, 131072, 64);
    cvt(O32, O16, BN * 512);
    launch_gemm(stream, O16, Wdaout, dq32, da_out_b, 2048, 512, 512, 512, 512, 512);
    pad16_k<<<(unsigned)(((long)B * 258 * 512 + 255) / 256), 256, 0, stream>>>(dq32, dqp, B, 256, 512);
    for (int t = 0; t < 3; ++t)       // conv1 (D->C, K=3) as 3 shifted GEMMs
      launch_gemm(stream, dqp + t * 512, Wdpc1[t], h1, t == 0 ? dp_c1_b : nullptr,
                  256, 256, 512, 512, 512, 256, B, 1, (long)258 * 512, 0, 0, 0,
                  (long)256 * 256, 0, 1.f, t > 0);
    zero_rows16_k<<<(B * 2 * 256 + 255) / 256, 256, 0, stream>>>(h1p, B, 256, 256);
    relu_ln_k<<<dim3(256, B), 256, 0, stream>>>(h1, dp_ln1_g, dp_ln1_b, h1p, nullptr, 256);
    for (int t = 0; t < 3; ++t)       // conv2 (C->C, K=3)
      launch_gemm(stream, h1p + t * 256, Wdpc2[t], h2, t == 0 ? dp_c2_b : nullptr,
                  256, 256, 256, 256, 256, 256, B, 1, (long)258 * 256, 0, 0, 0,
                  (long)256 * 256, 0, 1.f, t > 0);
    relu_ln_k<<<dim3(256, B), 256, 0, stream>>>(h2, dp_ln2_g, dp_ln2_b, nullptr, h2n, 256);
    rowdot_k<<<dim3((unsigned)(BN / 4)), dim3(32, 4), 0, stream>>>(h2n, dp_o_w, dp_o_b, draw, BN);
    dur_out_k<<<(int)(BN + 255) / 256, 256, 0, stream>>>(draw, Tg, out);
  }

  // =================== pitch branch (Nq = Tmax, zero-padded) ==============
  {
    cur = chk;                              // reuse duration scratch
    expand_k<<<dim3(Tmax, B), 256, 0, stream>>>(phon, cs, comb16, Tmax);

    float*     Q32  = A32(BT * 512);
    _Float16*  Q16  = A16(BT * 512);
    float*     KV32 = A32(BN * 1024);
    _Float16*  K16  = A16(BN * 512);
    _Float16*  Vt16 = A16(BN * 512);
    float*     S32  = A32((long)64 * 2048 * 256);
    _Float16*  P16  = A16((long)64 * 2048 * 256);
    float*     O32  = A32(BT * 512);
    _Float16*  O16  = A16(BT * 512);
    float*     pq32 = A32(BT * 512);
    _Float16*  pqp  = A16((long)B * 2050 * 512);
    float*     g1   = A32(BT * 256);
    _Float16*  g1p  = A16((long)B * 2050 * 256);
    float*     g2   = A32(BT * 256);
    float*     g2n  = A32(BT * 256);
    float*     pc32 = A32(BT * 512);

    // Q projection reads expanded cols [0,512) of combined (lda = 768)
    launch_gemm(stream, comb16, Wpaq, Q32, pa_in_b, 16384, 512, 512, 768, 512, 512);
    cvt(Q32, Q16, BT * 512);
    launch_gemm(stream, spk16, Wpakv, KV32, pa_in_b + 512, 2048, 1024, 512, 512, 512, 1024);
    slice16_k<<<(unsigned)((BN * 512 + 255) / 256), 256, 0, stream>>>(KV32, K16, BN, 512, 1024, 0);
    vtrans16_k<<<(unsigned)((BN * 512 + 255) / 256), 256, 0, stream>>>(KV32, Vt16, B, 256, 1024, 512);
    launch_gemm(stream, Q16, K16, S32, nullptr, 2048, 256, 64, 512, 512, 256,
                64, 8, 1048576, 64, 131072, 64, 4194304, 524288, 0.125f);
    softmax16_k<<<64 * 2048, 256, 0, stream>>>(S32, P16);
    launch_gemm(stream, P16, Vt16, O32, nullptr, 2048, 64, 256, 256, 256, 512,
                64, 8, 4194304, 524288, 131072, 16384, 1048576, 64);
    cvt(O32, O16, BT * 512);
    launch_gemm(stream, O16, Wpaout, pq32, pa_out_b, 16384, 512, 512, 512, 512, 512);
    pad16_k<<<(unsigned)(((long)B * 2050 * 512 + 255) / 256), 256, 0, stream>>>(pq32, pqp, B, 2048, 512);
    for (int t = 0; t < 3; ++t)
      launch_gemm(stream, pqp + t * 512, Wppc1[t], g1, t == 0 ? pp_c1_b : nullptr,
                  2048, 256, 512, 512, 512, 256, B, 1, (long)2050 * 512, 0, 0, 0,
                  (long)2048 * 256, 0, 1.f, t > 0);
    zero_rows16_k<<<(B * 2 * 256 + 255) / 256, 256, 0, stream>>>(g1p, B, 2048, 256);
    relu_ln_k<<<dim3(2048, B), 256, 0, stream>>>(g1, pp_ln1_g, pp_ln1_b, g1p, nullptr, 2048);
    for (int t = 0; t < 3; ++t)
      launch_gemm(stream, g1p + t * 256, Wppc2[t], g2, t == 0 ? pp_c2_b : nullptr,
                  2048, 256, 256, 256, 256, 256, B, 1, (long)2050 * 256, 0, 0, 0,
                  (long)2048 * 256, 0, 1.f, t > 0);
    relu_ln_k<<<dim3(2048, B), 256, 0, stream>>>(g2, pp_ln2_g, pp_ln2_b, nullptr, g2n, 2048);
    rowdot_k<<<dim3((unsigned)(BT / 4)), dim3(32, 4), 0, stream>>>(g2n, pp_o_w, pp_o_b, praw, BT);
    pitch_fill_k<<<dim3(Tmax, B), 256, 0, stream>>>(praw, pe_w, pe_b, comb16, out, Tg, Tmax);
    // final projection: combined[16384x768] x proj_w^T[512x768] -> p_c
    launch_gemm(stream, comb16, Wproj, pc32, proj_b, 16384, 512, 768, 768, 768, 512);
    pc_out_k<<<dim3(Tmax, B), 256, 0, stream>>>(pc32, Tg, out, Tmax);
  }
}
